// MaskedCrossAttention_11974368821815
// MI455X (gfx1250) — matrix operand order repaired and checked by reading, not compiled, not run on hardware
//
#include <hip/hip_runtime.h>

typedef __attribute__((ext_vector_type(16))) _Float16 v16h;
typedef __attribute__((ext_vector_type(8)))  _Float16 v8h;
typedef __attribute__((ext_vector_type(8)))  float    v8f;
typedef __attribute__((ext_vector_type(4)))  unsigned int v4u;
typedef __attribute__((ext_vector_type(8)))  int      v8i;
typedef __attribute__((ext_vector_type(4)))  int      v4i;

static constexpr int kB = 2, kTX = 1024, kTY = 4096, kC = 768, kH = 12, kD = 64;

#if defined(__has_builtin)
#  if __has_builtin(__builtin_amdgcn_tensor_load_to_lds)
#    define HAVE_TDM 1
#  endif
#  if __has_builtin(__builtin_amdgcn_s_wait_tensorcnt)
#    define WAIT_TENSOR(n) __builtin_amdgcn_s_wait_tensorcnt((short)(n))
#  endif
#endif
#ifndef HAVE_TDM
#  define HAVE_TDM 0
#endif
#ifndef WAIT_TENSOR
#  define WAIT_TENSOR(n) asm volatile("s_wait_tensorcnt %0" ::"i"(n))
#endif

__device__ __forceinline__ v8f wmma_f16(v16h a, v16h b, v8f c) {
  return __builtin_amdgcn_wmma_f32_16x16x32_f16(false, a, false, b, (short)0, c, false, false);
}

__device__ __forceinline__ v16h cat8(v8h lo, v8h hi) {
  return __builtin_shufflevector(lo, hi, 0,1,2,3,4,5,6,7,8,9,10,11,12,13,14,15);
}

#if HAVE_TDM
// Generic pointer to shared memory: LDS aperture occupies addr[63:32],
// addr[31:0] is the in-LDS byte offset (ISA 10.2 aperture rules).
__device__ __forceinline__ unsigned lds_byte_off(const void* p) {
  return (unsigned)(unsigned long long)(size_t)p;
}

// Build a D# for a 2D f16 tile load and fire the TDM.
__device__ __forceinline__ void tdm_load_2d_f16(
    const _Float16* tile_gaddr, unsigned lds_off,
    unsigned tile_d0, unsigned tile_d1,
    unsigned tensor_d0, unsigned tensor_d1,
    unsigned long long stride0_elems)
{
  unsigned long long ga = (unsigned long long)(size_t)tile_gaddr;
  v4u g0 = {};
  g0[0] = 1u;                                        // count=1 (valid user D#)
  g0[1] = lds_off;                                   // lds_addr [63:32]
  g0[2] = (unsigned)(ga & 0xffffffffu);              // global_addr[31:0]
  g0[3] = (unsigned)((ga >> 32) & 0x01ffffffu)       // global_addr[56:32]
        | (2u << 30);                                // type=2 ("image")
  v8i g1 = {};
  g1[0] = (int)(1u << 16);                           // data_size=1 (2B), wg_mask=0
  g1[1] = (int)((tensor_d0 & 0xffffu) << 16);        // tensor_dim0[15:0] @ [63:48]
  g1[2] = (int)(((tensor_d0 >> 16) & 0xffffu)
        | ((tensor_d1 & 0xffffu) << 16));            // dim0[31:16] | dim1[15:0]
  g1[3] = (int)(((tensor_d1 >> 16) & 0xffffu)
        | ((tile_d0 & 0xffffu) << 16));              // dim1[31:16] | tile_dim0
  g1[4] = (int)(tile_d1 & 0xffffu);                  // tile_dim1 (tile_dim2=0)
  g1[5] = (int)(stride0_elems & 0xffffffffu);        // tensor_dim0_stride[31:0]
  g1[6] = (int)((stride0_elems >> 32) & 0xffffu);    // stride0[47:32]
  g1[7] = 0;
  v4i z4 = {};
  v8i z8 = {};
  __builtin_amdgcn_tensor_load_to_lds(g0, g1, z4, z4, z8, 0);
}
#endif

// ---------------------------------------------------------------------------
// Tiled WMMA GEMM: C[M,N] = A[M,K] * W[K,N] (f16 operands, f32 accumulate).
// Block 256 threads = 8 waves (4M x 2N), tile 64x64, K step 32.
// Each wave computes 16x32 (2 WMMAs/K-step, A fragment reused).
// W tile is stored TRANSPOSED in LDS so B fragments are contiguous b128 reads.
// MODE 0: A=f32 x, epilogue RoPE(x_t) -> qh [B][H][TX][D] f16
// MODE 1: A=f32 y, epilogue n<C: RoPE(y_t) -> kh; n>=C: -> vT [B][H][D][TY]
// MODE 2: A=f16 attn, direct f32 store
// ---------------------------------------------------------------------------
template<int MODE>
__global__ __launch_bounds__(256) void gemm_rope(
    const float*    __restrict__ Af32,
    const _Float16* __restrict__ Af16,
    const float*    __restrict__ W,
    const float*    __restrict__ pos,
    const float*    __restrict__ invf,
    _Float16*       __restrict__ outA,
    _Float16*       __restrict__ outV,
    float*          __restrict__ outF,
    int M, int N, int K, int T)
{
  __shared__ __align__(16) _Float16 As [64][40];   // [m][k]
  __shared__ __align__(16) _Float16 Bst[64][40];   // [n][k]  (transposed)
  __shared__ __align__(16) float    Ct [64][66];

  const int tid  = threadIdx.x;
  const int wave = tid >> 5, lane = tid & 31;
  const int wm = wave >> 1, wn = wave & 1;         // 4 x 2 waves
  const int fm = lane & 15, fh = lane >> 4;
  const int m0 = blockIdx.x * 64, n0 = blockIdx.y * 64;

  const int arow = tid >> 2, acol = (tid & 3) * 8;     // A: 64x32, 8 elem/thread
  const int brow = tid >> 3, bcol = (tid & 7) * 8;     // B: 32x64, 8 elem/thread
  const int kb  = fh * 8;
  const int kb2 = fh * 16;

  v8f c0 = {}, c1 = {};
  for (int kc = 0; kc < K; kc += 32) {
    if (MODE == 2) {
      *(v8h*)&As[arow][acol] =
          *(const v8h*)(Af16 + (size_t)(m0 + arow) * K + kc + acol);
    } else {
      const float4 a0 = *(const float4*)(Af32 + (size_t)(m0 + arow) * K + kc + acol);
      const float4 a1 = *(const float4*)(Af32 + (size_t)(m0 + arow) * K + kc + acol + 4);
      v8h av;
      av[0] = (_Float16)a0.x; av[1] = (_Float16)a0.y;
      av[2] = (_Float16)a0.z; av[3] = (_Float16)a0.w;
      av[4] = (_Float16)a1.x; av[5] = (_Float16)a1.y;
      av[6] = (_Float16)a1.z; av[7] = (_Float16)a1.w;
      *(v8h*)&As[arow][acol] = av;
    }
    {
      const float4 b0 = *(const float4*)(W + (size_t)(kc + brow) * N + n0 + bcol);
      const float4 b1 = *(const float4*)(W + (size_t)(kc + brow) * N + n0 + bcol + 4);
      Bst[bcol + 0][brow] = (_Float16)b0.x;
      Bst[bcol + 1][brow] = (_Float16)b0.y;
      Bst[bcol + 2][brow] = (_Float16)b0.z;
      Bst[bcol + 3][brow] = (_Float16)b0.w;
      Bst[bcol + 4][brow] = (_Float16)b1.x;
      Bst[bcol + 5][brow] = (_Float16)b1.y;
      Bst[bcol + 6][brow] = (_Float16)b1.z;
      Bst[bcol + 7][brow] = (_Float16)b1.w;
    }
    if (kc + 32 < K)   // pull next W tile toward L2/WGP$ while we compute
      __builtin_prefetch(W + (size_t)(kc + 32 + brow) * N + n0 + bcol, 0, 1);
    __syncthreads();

    v16h a = cat8(*(const v8h*)&As[wm * 16 + fm][kb2],
                  *(const v8h*)&As[wm * 16 + fm][kb2 + 8]);
    const int bn0 = wn * 32 + fm;
    v16h b0f = cat8(*(const v8h*)&Bst[bn0][kb2],
                    *(const v8h*)&Bst[bn0][kb2 + 8]);
    v16h b1f = cat8(*(const v8h*)&Bst[bn0 + 16][kb2],
                    *(const v8h*)&Bst[bn0 + 16][kb2 + 8]);
    c0 = wmma_f16(a, b0f, c0);
    c1 = wmma_f16(a, b1f, c1);
    __syncthreads();
  }

  if (MODE == 2) {
    #pragma unroll
    for (int r = 0; r < 8; ++r) {
      int row = m0 + wm * 16 + r + 8 * fh;
      outF[(size_t)row * N + n0 + wn * 32 + fm]      = c0[r];
      outF[(size_t)row * N + n0 + wn * 32 + 16 + fm] = c1[r];
    }
    return;
  }

  #pragma unroll
  for (int r = 0; r < 8; ++r) {
    Ct[wm * 16 + r + 8 * fh][wn * 32 + fm]      = c0[r];
    Ct[wm * 16 + r + 8 * fh][wn * 32 + 16 + fm] = c1[r];
  }
  __syncthreads();

  #pragma unroll
  for (int it = 0; it < 8; ++it) {
    int idx = tid + it * 256;          // 2048 column-pairs in the 64x64 tile
    int row = idx >> 5;
    int pc  = (idx & 31) * 2;
    int gm = m0 + row, gn = n0 + pc;
    int b = gm / T, t = gm % T;
    float x1 = Ct[row][pc], x2 = Ct[row][pc + 1];
    if (MODE == 0 || gn < kC) {
      int hh = gn >> 6, d = gn & 63;
      float p = pos[b * T + t];
      float f = p * invf[d >> 1];
      float cs = __cosf(f), sn = __sinf(f);
      _Float16* dst = outA + ((size_t)(b * kH + hh) * T + t) * kD + d;
      dst[0] = (_Float16)(x1 * cs - x2 * sn);
      dst[1] = (_Float16)(x2 * cs + x1 * sn);
    } else {
      int n2 = gn - kC;
      int hh = n2 >> 6, d = n2 & 63;
      _Float16* dst = outV + ((size_t)(b * kH + hh) * kD + d) * kTY + t;
      dst[0]   = (_Float16)x1;
      dst[kTY] = (_Float16)x2;
    }
  }
}

// ---------------------------------------------------------------------------
// Flash attention. Block = 4 waves = 64 queries of one (b,h).
// K/V tiles (shared by all 4 waves) are DMA'd into LDS by the Tensor Data
// Mover (double-buffered, s_wait_tensorcnt pipelined); waves read WMMA
// fragments from LDS and run online softmax between the QK^T and PV WMMAs.
// ---------------------------------------------------------------------------
__global__ __launch_bounds__(128) void attn_fwd(
    const _Float16* __restrict__ qh,
    const _Float16* __restrict__ kh,
    const _Float16* __restrict__ vT,
    const float*    __restrict__ x_t,
    const float*    __restrict__ y_t,
    const int*      __restrict__ dist_p,
    const int*      __restrict__ mind_p,
    _Float16*       __restrict__ attnO)
{
  __shared__ __align__(64) _Float16 Ks[2][32][64];   // [buf][key][d]
  __shared__ __align__(64) _Float16 Vs[2][64][32];   // [buf][d][key]
  __shared__ __align__(16) _Float16 Pst[4][16][40];  // per-wave P transpose

  const int wave = threadIdx.x >> 5, lane = threadIdx.x & 31;
  const int fm = lane & 15, fh = lane >> 4;
  const int qBase = blockIdx.x * 64 + wave * 16;
  const int h = blockIdx.y, b = blockIdx.z;
  const float dist = (float)dist_p[0];
  const float mind = (float)mind_p[0];
  const size_t bh = (size_t)b * kH + h;
  const int kb = fh * 8, kb2 = fh * 16;

  const _Float16* qrow = qh + (bh * kTX + qBase + fm) * (size_t)kD;
  v16h q0 = cat8(*(const v8h*)(qrow + kb2),      *(const v8h*)(qrow + kb2 + 8));
  v16h q1 = cat8(*(const v8h*)(qrow + kb2 + 32), *(const v8h*)(qrow + kb2 + 40));

  float xt[8], mrow[8], lrow[8];
  #pragma unroll
  for (int r = 0; r < 8; ++r) {
    xt[r] = x_t[b * kTX + qBase + r + 8 * fh] - mind;
    mrow[r] = -1e30f;
    lrow[r] = 0.f;
  }
  v8f acc[4] = {};

  const _Float16* kbase = kh + bh * (size_t)kTY * kD;
  const _Float16* vbase = vT + bh * (size_t)kD * kTY;
  const int nIter = kTY / 32;
  int cur = 0;

#if HAVE_TDM
  if (wave == 0) {
    // K tile: tensor 64(d) x TY(key), stride 64; tile 64 x 32 keys
    tdm_load_2d_f16(kbase, lds_byte_off(&Ks[0][0][0]), 64, 32, 64, kTY, 64);
    // V tile: tensor TY(t) x 64(d), stride TY; tile 32 keys x 64 d rows
    tdm_load_2d_f16(vbase, lds_byte_off(&Vs[0][0][0]), 32, 64, kTY, 64, kTY);
  }
#endif

  for (int it = 0; it < nIter; ++it) {
    const int k0 = it * 32;
#if HAVE_TDM
    if (wave == 0) {
      if (it + 1 < nIter) {            // prefetch next tile into other buffer
        const int k1 = k0 + 32;
        tdm_load_2d_f16(kbase + (size_t)k1 * kD, lds_byte_off(&Ks[cur ^ 1][0][0]),
                        64, 32, 64, kTY, 64);
        tdm_load_2d_f16(vbase + k1, lds_byte_off(&Vs[cur ^ 1][0][0]),
                        32, 64, kTY, 64, kTY);
        WAIT_TENSOR(2);                // current tile's 2 DMAs done; next 2 fly
      } else {
        WAIT_TENSOR(0);
      }
    }
    __syncthreads();
#else
    {   // fallback: cooperative 32B-chunk copy (256 chunks / 128 threads)
      #pragma unroll
      for (int cc = 0; cc < 2; ++cc) {
        int ch = threadIdx.x + cc * 128;
        if (ch < 128) {
          int key = ch >> 2, dd = (ch & 3) * 16;
          *(v16h*)&Ks[cur][key][dd] = *(const v16h*)(kbase + (size_t)(k0 + key) * kD + dd);
        } else {
          int c2 = ch - 128;
          int d = c2 >> 1, kk = (c2 & 1) * 16;
          *(v16h*)&Vs[cur][d][kk] = *(const v16h*)(vbase + (size_t)d * kTY + k0 + kk);
        }
      }
    }
    __syncthreads();
#endif

    // S = Q K^T for 16 queries x 32 keys (4 WMMAs), from LDS tiles
    const _Float16* kr0 = &Ks[cur][fm][kb2];
    const _Float16* kr1 = &Ks[cur][16 + fm][kb2];
    v16h bk0a = *(const v16h*)(kr0);
    v16h bk0b = *(const v16h*)(kr0 + 32);
    v16h bk1a = *(const v16h*)(kr1);
    v16h bk1b = *(const v16h*)(kr1 + 32);

    v8f s0 = {}, s1 = {};
    s0 = wmma_f16(q0, bk0a, s0);
    s0 = wmma_f16(q1, bk0b, s0);
    s1 = wmma_f16(q0, bk1a, s1);
    s1 = wmma_f16(q1, bk1b, s1);

    float yt0 = y_t[b * kTY + k0 + fm];
    float yt1 = y_t[b * kTY + k0 + 16 + fm];

    #pragma unroll
    for (int r = 0; r < 8; ++r) {
      float v0 = s0[r] * 0.125f;
      float v1 = s1[r] * 0.125f;
      bool ok0 = (xt[r] >= yt0) && (xt[r] <= yt0 + dist);
      bool ok1 = (xt[r] >= yt1) && (xt[r] <= yt1 + dist);
      v0 = ok0 ? v0 : -1e30f;
      v1 = ok1 ? v1 : -1e30f;
      float rmax = fmaxf(v0, v1);
      #pragma unroll
      for (int s = 1; s < 16; s <<= 1) rmax = fmaxf(rmax, __shfl_xor(rmax, s, 32));
      float mn = fmaxf(mrow[r], rmax);
      float alpha = __expf(mrow[r] - mn);
      float p0 = ok0 ? __expf(v0 - mn) : 0.f;
      float p1 = ok1 ? __expf(v1 - mn) : 0.f;
      float rs = p0 + p1;
      #pragma unroll
      for (int s = 1; s < 16; s <<= 1) rs += __shfl_xor(rs, s, 32);
      lrow[r] = lrow[r] * alpha + rs;
      mrow[r] = mn;
      #pragma unroll
      for (int dt = 0; dt < 4; ++dt) acc[dt][r] *= alpha;
      const int mr = r + 8 * fh;
      Pst[wave][mr][fm]      = (_Float16)p0;
      Pst[wave][mr][16 + fm] = (_Float16)p1;
    }
    __syncthreads();   // C layout (lane=key) -> A layout (lane=query)
    v16h pf = cat8(*(const v8h*)&Pst[wave][fm][kb2],
                   *(const v8h*)&Pst[wave][fm][kb2 + 8]);
    #pragma unroll
    for (int dt = 0; dt < 4; ++dt) {
      v16h bv = *(const v16h*)&Vs[cur][dt * 16 + fm][kb2];
      acc[dt] = wmma_f16(pf, bv, acc[dt]);
    }
    __syncthreads();
    cur ^= 1;
  }

  #pragma unroll
  for (int r = 0; r < 8; ++r) {
    float inv = lrow[r] > 0.f ? 1.f / lrow[r] : 0.f;
    int qr = qBase + r + 8 * fh;
    #pragma unroll
    for (int dt = 0; dt < 4; ++dt) {
      attnO[((size_t)b * kTX + qr) * kC + h * kD + dt * 16 + fm] =
          (_Float16)(acc[dt][r] * inv);
    }
  }
}

// ---------------------------------------------------------------------------
extern "C" void kernel_launch(void* const* d_in, const int* in_sizes, int n_in,
                              void* d_out, int out_size, void* d_ws, size_t ws_size,
                              hipStream_t stream) {
  (void)in_sizes; (void)n_in; (void)out_size; (void)ws_size;
  const float* x    = (const float*)d_in[0];
  const float* x_t  = (const float*)d_in[1];
  const float* y    = (const float*)d_in[2];
  const float* y_t  = (const float*)d_in[3];
  const int*   dist = (const int*)d_in[4];
  const int*   mind = (const int*)d_in[5];
  const float* Wq   = (const float*)d_in[6];
  const float* Wkv  = (const float*)d_in[7];
  const float* Wp   = (const float*)d_in[8];
  const float* invf = (const float*)d_in[9];
  float* out = (float*)d_out;

  char* ws = (char*)d_ws;
  _Float16* qh  = (_Float16*)(ws);
  _Float16* kh  = (_Float16*)(ws + 3145728ull);
  _Float16* vT  = (_Float16*)(ws + 3145728ull + 12582912ull);
  _Float16* att = (_Float16*)(ws + 3145728ull + 2ull * 12582912ull);

  gemm_rope<0><<<dim3(32, 12), 256, 0, stream>>>(
      x, nullptr, Wq, x_t, invf, qh, nullptr, nullptr, 2048, 768, 768, kTX);
  gemm_rope<1><<<dim3(128, 24), 256, 0, stream>>>(
      y, nullptr, Wkv, y_t, invf, kh, vT, nullptr, 8192, 1536, 768, kTY);
  attn_fwd<<<dim3(16, 12, 2), 128, 0, stream>>>(
      qh, kh, vT, x_t, y_t, dist, mind, att);
  gemm_rope<2><<<dim3(32, 12), 256, 0, stream>>>(
      nullptr, att, Wp, nullptr, nullptr, nullptr, nullptr, out, 2048, 768, 768, kTX);
}
